// SupConLoss_7911329759548
// MI455X (gfx1250) — compile-verified, matrix operand order checked by reading, not run on hardware
//
#include <hip/hip_runtime.h>
#include <hip/hip_bf16.h>
#include <math.h>

typedef __attribute__((ext_vector_type(16))) _Float16 v16h;
typedef __attribute__((ext_vector_type(4)))  _Float16 v4h;
typedef __attribute__((ext_vector_type(8)))  float    v8f;

#define D_DIM    128
#define TAU_INV  (1.0f / 0.07f)
// exp(x/tau) == exp2(x * TAU_INV * log2(e)) : fold both scales into one mul
#define EXP2_SCALE (TAU_INV * 1.44269504088896340736f)
#define LN2F 0.69314718055994530942f

// Raw v_exp_f32 (exp2) without libm's denormal-range fixup: our arguments are
// bounded by |sim/tau * log2e| <= ~20.7, far from the -126 underflow edge.
__device__ __forceinline__ float fast_exp2(float x) {
#if __has_builtin(__builtin_amdgcn_exp2f)
  return __builtin_amdgcn_exp2f(x);
#else
  return exp2f(x);
#endif
}
__device__ __forceinline__ float fast_log2(float x) {
#if __has_builtin(__builtin_amdgcn_logf)
  return __builtin_amdgcn_logf(x);
#else
  return log2f(x);
#endif
}

// ---------------------------------------------------------------------------
// Kernel 1: L2-normalize each row (fp32 in, f16 out) and zero accumulators.
// One wave (32 lanes) per row, 4 floats per lane.
// ---------------------------------------------------------------------------
__global__ void __launch_bounds__(256)
supcon_normalize(const float* __restrict__ feats,
                 _Float16* __restrict__ f16,
                 float* __restrict__ denom,
                 float* __restrict__ pos,
                 int N) {
  int wave = threadIdx.x >> 5;
  int lane = threadIdx.x & 31;
  int row  = blockIdx.x * 8 + wave;
  if (row >= N) return;

  const float4* src = (const float4*)(feats + (size_t)row * D_DIM);
  float4 v = src[lane];
  float s = v.x * v.x + v.y * v.y + v.z * v.z + v.w * v.w;
#pragma unroll
  for (int m = 16; m >= 1; m >>= 1) s += __shfl_xor(s, m, 32);
  float inv = 1.0f / fmaxf(sqrtf(s), 1e-8f);

  v4h h;
  h[0] = (_Float16)(v.x * inv);
  h[1] = (_Float16)(v.y * inv);
  h[2] = (_Float16)(v.z * inv);
  h[3] = (_Float16)(v.w * inv);
  ((v4h*)(f16 + (size_t)row * D_DIM))[lane] = h;

  if (lane == 0) { denom[row] = 0.0f; pos[row] = 0.0f; }
}

// ---------------------------------------------------------------------------
// Kernel 2: fused Gram-matrix (WMMA f16->f32) + exp2 + masked row sums.
// Block = 256 threads (8 waves). Each wave owns a 16-row strip (A frags in
// registers, K=128 as 4x K32) and sweeps column tiles TWO at a time with two
// independent accumulator chains so consecutive v_wmma ops have no RAW dep.
// C layout (16x16 f32): slot(vgpr r, lane l): M = r + 8*(l>=16), N = l&15.
// A/B fragments use the identical (half,slot)->K map => permutation-safe.
// ---------------------------------------------------------------------------
__global__ void __launch_bounds__(256)
supcon_tiles(const _Float16* __restrict__ f16,
             const int* __restrict__ labels,
             float* __restrict__ denom,
             float* __restrict__ pos,
             int N, int colTiles) {
  const int lane = threadIdx.x & 31;
  const int wave = threadIdx.x >> 5;
  const int hi   = lane >> 4;      // half-wave: selects K+16 within each K32 chunk
  const int lm   = lane & 15;      // M index for A rows / N index for B cols

  const int rowBase  = blockIdx.x * 128 + wave * 16;
  const int colChunk = blockIdx.y * (colTiles * 16);

  // A fragments: 16 rows x 128 K, register-resident for the whole sweep.
  const _Float16* arow = f16 + (size_t)(rowBase + lm) * D_DIM + hi * 16;
  const v16h a0 = *(const v16h*)(arow +  0);
  const v16h a1 = *(const v16h*)(arow + 32);
  const v16h a2 = *(const v16h*)(arow + 64);
  const v16h a3 = *(const v16h*)(arow + 96);

  int   rowLab[8];
  float dAcc[8], pAcc[8];
#pragma unroll
  for (int r = 0; r < 8; ++r) {
    rowLab[r] = labels[rowBase + hi * 8 + r];
    dAcc[r] = 0.0f;
    pAcc[r] = 0.0f;
  }

  for (int ct = 0; ct < colTiles; ct += 2) {
    const int colBase0 = colChunk + ct * 16;
    const int colBase1 = colBase0 + 16;

    const _Float16* brow0 = f16 + (size_t)(colBase0 + lm) * D_DIM + hi * 16;
    const _Float16* brow1 = brow0 + 16 * D_DIM;
    // prefetch the next pair (gfx1250 global_prefetch_b8)
    __builtin_prefetch(brow0 + 32 * D_DIM, 0, 3);

    // Issue all 8 B-tile loads up front; wmma waits retire incrementally.
    const v16h b00 = *(const v16h*)(brow0 +  0);
    const v16h b01 = *(const v16h*)(brow0 + 32);
    const v16h b02 = *(const v16h*)(brow0 + 64);
    const v16h b03 = *(const v16h*)(brow0 + 96);
    const v16h b10 = *(const v16h*)(brow1 +  0);
    const v16h b11 = *(const v16h*)(brow1 + 32);
    const v16h b12 = *(const v16h*)(brow1 + 64);
    const v16h b13 = *(const v16h*)(brow1 + 96);

    const int labJ0 = labels[colBase0 + lm];
    const int labJ1 = labels[colBase1 + lm];

    v8f c0 = {};
    v8f c1 = {};
    // Interleaved: consecutive wmmas target different accumulators (no RAW).
    c0 = __builtin_amdgcn_wmma_f32_16x16x32_f16(false, a0, false, b00, (short)0, c0, false, false);
    c1 = __builtin_amdgcn_wmma_f32_16x16x32_f16(false, a0, false, b10, (short)0, c1, false, false);
    c0 = __builtin_amdgcn_wmma_f32_16x16x32_f16(false, a1, false, b01, (short)0, c0, false, false);
    c1 = __builtin_amdgcn_wmma_f32_16x16x32_f16(false, a1, false, b11, (short)0, c1, false, false);
    c0 = __builtin_amdgcn_wmma_f32_16x16x32_f16(false, a2, false, b02, (short)0, c0, false, false);
    c1 = __builtin_amdgcn_wmma_f32_16x16x32_f16(false, a2, false, b12, (short)0, c1, false, false);
    c0 = __builtin_amdgcn_wmma_f32_16x16x32_f16(false, a3, false, b03, (short)0, c0, false, false);
    c1 = __builtin_amdgcn_wmma_f32_16x16x32_f16(false, a3, false, b13, (short)0, c1, false, false);

    // Diagonal exists only in the tile whose columns equal this wave's rows —
    // wave-uniform test, so the common path carries no per-lane cmp/cndmask.
    const bool diag0 = (colBase0 == rowBase);
    const bool diag1 = (colBase1 == rowBase);
    const int  mSlot = lm - hi * 8;   // r index that hits the diagonal (if 0..7)

#pragma unroll
    for (int r = 0; r < 8; ++r) {
      float e0 = fast_exp2(c0[r] * EXP2_SCALE);
      float e1 = fast_exp2(c1[r] * EXP2_SCALE);
      if (diag0 && r == mSlot) e0 = 0.0f;
      if (diag1 && r == mSlot) e1 = 0.0f;
      dAcc[r] += e0 + e1;
      pAcc[r] += (rowLab[r] == labJ0 ? e0 : 0.0f) + (rowLab[r] == labJ1 ? e1 : 0.0f);
    }
  }

  // Reduce each row's 16 column-partials across the 16 lanes of each half-wave
  // (xor masks 1,2,4,8 never cross the lane-16 boundary).
#pragma unroll
  for (int r = 0; r < 8; ++r) {
    float d = dAcc[r], p = pAcc[r];
#pragma unroll
    for (int m = 8; m >= 1; m >>= 1) {
      d += __shfl_xor(d, m, 32);
      p += __shfl_xor(p, m, 32);
    }
    if (lm == 0) {
      const int i = rowBase + hi * 8 + r;
      atomicAdd(&denom[i], d);
      atomicAdd(&pos[i], p);
    }
  }
}

// ---------------------------------------------------------------------------
// Kernel 3: per-row loss + mean over valid rows (single block).
// loss_i = -ln((pos+1e-12)/max(denom,1e-8))
//        = ln2 * (log2(max(denom,1e-8)) - log2(pos+1e-12))
// ---------------------------------------------------------------------------
__global__ void __launch_bounds__(256)
supcon_finalize(const float* __restrict__ denom,
                const float* __restrict__ pos,
                float* __restrict__ out, int N) {
  __shared__ float sSum[256];
  __shared__ int   sCnt[256];
  float sum = 0.0f;
  int   cnt = 0;
  for (int i = threadIdx.x; i < N; i += 256) {
    float p = pos[i];
    float d = fmaxf(denom[i], 1e-8f);
    if (p > 0.0f) {                 // valid: at least one positive pair
      sum += LN2F * (fast_log2(d) - fast_log2(p + 1e-12f));
      cnt += 1;
    }
  }
  sSum[threadIdx.x] = sum;
  sCnt[threadIdx.x] = cnt;
  __syncthreads();
  for (int s = 128; s > 0; s >>= 1) {
    if ((int)threadIdx.x < s) {
      sSum[threadIdx.x] += sSum[threadIdx.x + s];
      sCnt[threadIdx.x] += sCnt[threadIdx.x + s];
    }
    __syncthreads();
  }
  if (threadIdx.x == 0)
    out[0] = (sCnt[0] > 0) ? sSum[0] / (float)sCnt[0] : 0.0f;
}

// ---------------------------------------------------------------------------
extern "C" void kernel_launch(void* const* d_in, const int* in_sizes, int n_in,
                              void* d_out, int out_size, void* d_ws, size_t ws_size,
                              hipStream_t stream) {
  const float* feats  = (const float*)d_in[0];
  const int*   labels = (const int*)d_in[1];
  const int    N      = in_sizes[1];          // 8192

  char* ws = (char*)d_ws;
  _Float16* f16 = (_Float16*)ws;
  size_t f16Bytes = (size_t)N * D_DIM * sizeof(_Float16);
  f16Bytes = (f16Bytes + 255) & ~(size_t)255;
  float* denom = (float*)(ws + f16Bytes);
  float* pos   = denom + N;
  float* outF  = (float*)d_out;

  supcon_normalize<<<(N + 7) / 8, 256, 0, stream>>>(feats, f16, denom, pos, N);

  const int colTiles = 32;                    // 512 columns per chunk
  dim3 grid(N / 128, N / (colTiles * 16));    // (64, 16) -> 1024 blocks, 8192 waves
  supcon_tiles<<<grid, 256, 0, stream>>>(f16, labels, denom, pos, N, colTiles);

  supcon_finalize<<<1, 256, 0, stream>>>(denom, pos, outF, N);
}
